// Fusion_30872224923926
// MI455X (gfx1250) — matrix operand order repaired and checked by reading, not compiled, not run on hardware
//
#include <hip/hip_runtime.h>
#include <cstddef>

#define KN 256
#define NHALF 128   // output columns staged per workgroup (64KB LDS of W)

typedef __attribute__((ext_vector_type(16))) __bf16 v16bf;
typedef __attribute__((ext_vector_type(8)))  float  v8f;
typedef unsigned int u32x4 __attribute__((ext_vector_type(4)));
typedef int          i32x4 __attribute__((ext_vector_type(4)));
typedef int          i32x8 __attribute__((ext_vector_type(8)));

// Tensor Data Mover availability (per-pass: evaluates false on host pass)
#if defined(__has_builtin)
#if __has_builtin(__builtin_amdgcn_tensor_load_to_lds) && \
    __has_builtin(__builtin_amdgcn_s_wait_tensorcnt)
#define USE_TDM 1
#endif
#endif
#ifndef USE_TDM
#define USE_TDM 0
#endif

// ---------- helpers ----------
__device__ __forceinline__ unsigned f2ord(float f) {
  unsigned u = __float_as_uint(f);
  return (u & 0x80000000u) ? ~u : (u | 0x80000000u);   // monotone float->uint
}
__device__ __forceinline__ float ord2f(unsigned u) {
  return (u & 0x80000000u) ? __uint_as_float(u & 0x7FFFFFFFu)
                           : __uint_as_float(~u);
}
__device__ __forceinline__ unsigned short f2bf(float f) {  // RNE f32->bf16
  unsigned x = __float_as_uint(f);
  unsigned r = ((x >> 16) & 1u) + 0x7FFFu;
  return (unsigned short)((x + r) >> 16);
}

// ---------- fp32 -> bf16 staging (virtual concat of two row blocks) ----------
__global__ void cvt_concat_bf16(unsigned short* __restrict__ dst,
                                const float* __restrict__ A, int nA,
                                const float* __restrict__ B, size_t total) {
  size_t i = (size_t)blockIdx.x * blockDim.x + threadIdx.x;
  if (i >= total) return;
  size_t split = (size_t)nA * KN;
  float v = (i < split) ? A[i] : B[i - split];
  dst[i] = f2bf(v);
}

// ---------- z = h @ W^T via v_wmma_f32_16x16x32_bf16 ----------
// gridDim.y = 2: workgroup ny owns output cols [ny*128, ny*128+128).
// W rows for those cols (128 x 256 bf16 = 64KB) are staged in LDS once per
// workgroup (TDM DMA when available), then shared by all 8 waves.
// One wave computes a 16-row x 128-col strip of z (8 accumulators).
// A fragment (16x32 bf16): lane lr=l&15 -> row M=lr; half hi=l>>4;
//   VGPR v holds K pair at: hi*16 + v*2  (pairs contiguous)
// B fragment (32x16 bf16): lane -> col N=lr; VGPR v holds K = hi*16 + 2v, +1.
// C/D (16x16 f32): VGPR v -> M = v + hi*8, N = lr.
__global__ __launch_bounds__(256)
void gemm_bf16_wmma(float* __restrict__ z,
                    const unsigned short* __restrict__ hbf,
                    const unsigned short* __restrict__ wbf, int nRows) {
  __shared__ unsigned short wlds[NHALF * KN];           // 64 KB
  const int ny = blockIdx.y;
  const unsigned short* wsrc = wbf + (size_t)ny * NHALF * KN;

#if USE_TDM
  if ((threadIdx.x >> 5) == 0) {   // one wave issues the DMA (wave-level op)
    unsigned ldsoff = (unsigned)(unsigned long long)
        (__attribute__((address_space(3))) unsigned short*)wlds;
    unsigned long long ga = (unsigned long long)wsrc;
    // D# group0: count=1 | lds_addr | global_addr[56:0] | type=2
    u32x4 g0 = { 1u, ldsoff, (unsigned)ga,
                 (unsigned)((ga >> 32) & 0x01FFFFFFu) | 0x80000000u };
    // D# group1: data_size=2B; tensor_dim0=256, tensor_dim1=128;
    // tile_dim0=256, tile_dim1=128; tensor_dim0_stride=256
    i32x8 g1 = { 0x00010000,   // wg_mask=0 | data_size=1(2B)
                 0x01000000,   // tensor_dim0[15:0]=256 in [31:16]
                 0x00800000,   // tensor_dim0[31:16]=0 | tensor_dim1[15:0]=128
                 0x01000000,   // tensor_dim1[31:16]=0 | tile_dim0=256
                 0x00000080,   // tile_dim1=128 | tile_dim2=0
                 0x00000100,   // tensor_dim0_stride[31:0]=256
                 0, 0 };       // stride hi / tensor_dim1_stride = 0
    i32x4 gz = { 0, 0, 0, 0 };
#if __clang_major__ >= 23
    i32x8 gz8 = { 0, 0, 0, 0, 0, 0, 0, 0 };
    __builtin_amdgcn_tensor_load_to_lds(g0, g1, gz, gz, gz8, 0);
#else
    __builtin_amdgcn_tensor_load_to_lds(g0, g1, gz, gz, 0);
#endif
    __builtin_amdgcn_s_wait_tensorcnt(0);
  }
#else
  {
    const uint4* s4 = (const uint4*)wsrc;
    uint4* d4 = (uint4*)wlds;
    for (int i = threadIdx.x; i < NHALF * KN / 8; i += blockDim.x)
      d4[i] = s4[i];
  }
#endif
  __syncthreads();   // before any wave can exit

  int wave = blockIdx.x * (blockDim.x >> 5) + (threadIdx.x >> 5);
  int lane = threadIdx.x & 31;
  int row0 = wave * 16;
  if (row0 + 16 > nRows) return;   // whole-wave guard: EXEC stays all-1s
  int lr = lane & 15, hi = lane >> 4;

  v8f zero = {0.f, 0.f, 0.f, 0.f, 0.f, 0.f, 0.f, 0.f};
  v8f acc[8];
#pragma unroll
  for (int t = 0; t < 8; ++t) acc[t] = zero;

  union Frag { v16bf v; unsigned u[8]; };
  Frag Af, Bf;
  const unsigned short* hrow = hbf + (size_t)(row0 + lr) * KN;

  for (int kb = 0; kb < KN; kb += 32) {
#pragma unroll
    for (int v = 0; v < 8; ++v) {
      int koff = hi * 16 + v * 2;
      Af.u[v] = *(const unsigned*)(hrow + kb + koff);
    }
#pragma unroll
    for (int t = 0; t < 8; ++t) {
      const unsigned short* wrow = &wlds[(t * 16 + lr) * KN + kb + hi * 16];
#pragma unroll
      for (int v = 0; v < 8; ++v) Bf.u[v] = *(const unsigned*)(wrow + v * 2);
      acc[t] = __builtin_amdgcn_wmma_f32_16x16x32_bf16(
          false, Af.v, false, Bf.v, (short)0, acc[t], false, false);
    }
  }
#pragma unroll
  for (int t = 0; t < 8; ++t)
#pragma unroll
    for (int v = 0; v < 8; ++v)
      z[(size_t)(row0 + v + hi * 8) * KN + ny * NHALF + t * 16 + lr] =
          acc[t][v];
}

// ---------- s1[n]=z[n]·a[:256], s2[n]=z[n]·a[256:] (wave per row) ----------
__global__ void row_dots(float* __restrict__ s1, float* __restrict__ s2,
                         const float* __restrict__ z,
                         const float* __restrict__ a, int n) {
  int row = blockIdx.x * (blockDim.x >> 5) + (threadIdx.x >> 5);
  int lane = threadIdx.x & 31;
  if (row >= n) return;
  const float* zr = z + (size_t)row * KN;
  float d1 = 0.f, d2 = 0.f;
  for (int c = lane; c < KN; c += 32) {
    float zv = zr[c];
    d1 += zv * a[c];
    d2 += zv * a[KN + c];
  }
  for (int off = 16; off > 0; off >>= 1) {
    d1 += __shfl_down(d1, off);
    d2 += __shfl_down(d2, off);
  }
  if (lane == 0) { s1[row] = d1; s2[row] = d2; }
}

// ---------- per-layer state init ----------
__global__ void init_layer(unsigned* __restrict__ m_ord,
                           float* __restrict__ denom,
                           float* __restrict__ acc, int n) {
  size_t i = (size_t)blockIdx.x * blockDim.x + threadIdx.x;
  if (i < (size_t)n) { m_ord[i] = 0u; denom[i] = 0.f; }  // 0 < f2ord(-inf)
  if (i < (size_t)n * KN) acc[i] = 0.f;
}

// ---------- edge passes ----------
__global__ void edge_max(const int* __restrict__ src, const int* __restrict__ dst,
                         const float* __restrict__ s1, const float* __restrict__ s2,
                         unsigned* __restrict__ m_ord, int nE) {
  int i = blockIdx.x * blockDim.x + threadIdx.x;
  if (i >= nE) return;
  int d = dst[i];
  atomicMax(&m_ord[d], f2ord(s1[src[i]] + s2[d]));
}

__global__ void edge_sum(const int* __restrict__ src, const int* __restrict__ dst,
                         const float* __restrict__ s1, const float* __restrict__ s2,
                         const unsigned* __restrict__ m_ord,
                         float* __restrict__ denom, int nE) {
  int i = blockIdx.x * blockDim.x + threadIdx.x;
  if (i >= nE) return;
  int d = dst[i];
  float e = s1[src[i]] + s2[d];
  atomicAdd(&denom[d], __expf(e - ord2f(m_ord[d])));
}

__global__ void edge_scatter(const int* __restrict__ src, const int* __restrict__ dst,
                             const float* __restrict__ s1, const float* __restrict__ s2,
                             const unsigned* __restrict__ m_ord,
                             const float* __restrict__ denom,
                             const float* __restrict__ z,
                             float* __restrict__ acc, int nE) {
  int edge = blockIdx.x * (blockDim.x >> 5) + (threadIdx.x >> 5);
  int lane = threadIdx.x & 31;
  if (edge >= nE) return;
  int s = src[edge], d = dst[edge];
  float e = s1[s] + s2[d];
  float alpha = __expf(e - ord2f(m_ord[d])) / denom[d];
  const float* zs = z + (size_t)s * KN;
  float* ad = acc + (size_t)d * KN;
  for (int c = lane; c < KN; c += 32) atomicAdd(&ad[c], alpha * zs[c]);
}

// ---------- out = base + accPart (elementwise) ----------
__global__ void add_rows(float* __restrict__ out, const float* __restrict__ base,
                         const float* __restrict__ accPart, size_t total) {
  size_t i = (size_t)blockIdx.x * blockDim.x + threadIdx.x;
  if (i >= total) return;
  out[i] = base[i] + accPart[i];
}

// ---------- exer_out = emb + p1*B + p2*C, (p1,p2)=softmax(s1,s2) ----------
__global__ void exer_final(float* __restrict__ out, const float* __restrict__ emb,
                           const float* __restrict__ B, const float* __restrict__ C,
                           const float* __restrict__ e1w, const float* __restrict__ e1b,
                           const float* __restrict__ e2w, const float* __restrict__ e2b,
                           int n) {
  int row = blockIdx.x * (blockDim.x >> 5) + (threadIdx.x >> 5);
  int lane = threadIdx.x & 31;
  if (row >= n) return;
  const float* er = emb + (size_t)row * KN;
  const float* Br = B + (size_t)row * KN;
  const float* Cr = C + (size_t)row * KN;
  float d1 = 0.f, d2 = 0.f;
  for (int c = lane; c < KN; c += 32) {
    float ev = er[c];
    d1 += ev * e1w[c] + Br[c] * e1w[KN + c];
    d2 += ev * e2w[c] + Cr[c] * e2w[KN + c];
  }
  for (int off = 16; off > 0; off >>= 1) {
    d1 += __shfl_down(d1, off);
    d2 += __shfl_down(d2, off);
  }
  d1 = __shfl(d1, 0);
  d2 = __shfl(d2, 0);
  float sA = d1 + e1b[0], sB = d2 + e2b[0];
  float mx = fmaxf(sA, sB);
  float p1 = __expf(sA - mx), p2 = __expf(sB - mx);
  float inv = 1.f / (p1 + p2);
  p1 *= inv; p2 *= inv;
  for (int c = lane; c < KN; c += 32)
    out[(size_t)row * KN + c] = er[c] + p1 * Br[c] + p2 * Cr[c];
}

extern "C" void kernel_launch(void* const* d_in, const int* in_sizes, int n_in,
                              void* d_out, int out_size, void* d_ws, size_t ws_size,
                              hipStream_t stream) {
  (void)n_in; (void)out_size; (void)ws_size;
  const float* kn_emb   = (const float*)d_in[0];
  const float* exer_emb = (const float*)d_in[1];
  const float* stu_emb  = (const float*)d_in[2];
  const int* src_ke = (const int*)d_in[3];  const int* dst_ke = (const int*)d_in[4];
  const int* src_ek = (const int*)d_in[5];  const int* dst_ek = (const int*)d_in[6];
  const int* src_se = (const int*)d_in[7];  const int* dst_se = (const int*)d_in[8];
  const int* src_es = (const int*)d_in[9];  const int* dst_es = (const int*)d_in[10];
  const float* W_ke = (const float*)d_in[11]; const float* a_ke = (const float*)d_in[12];
  const float* W_ek = (const float*)d_in[13]; const float* a_ek = (const float*)d_in[14];
  const float* W_se = (const float*)d_in[15]; const float* a_se = (const float*)d_in[16];
  const float* W_es = (const float*)d_in[17]; const float* a_es = (const float*)d_in[18];
  // d_in[19..20] = k3_w/k3_b: softmax over a length-1 axis == 1.0, params are dead.
  const float* e1_w = (const float*)d_in[21]; const float* e1_b = (const float*)d_in[22];
  const float* e2_w = (const float*)d_in[23]; const float* e2_b = (const float*)d_in[24];

  const int EXER = in_sizes[1] / KN;
  const int STU  = in_sizes[2] / KN;
  const int nEK  = EXER + KN;        // 20256 (mult of 16)
  const int nES  = EXER + STU;       // 70000 (mult of 16)
  const int nMax = (nES > nEK) ? nES : nEK;
  const int E_KE = in_sizes[3], E_SE = in_sizes[7];

  char* wsp = (char*)d_ws; size_t off = 0;
  auto take = [&](size_t bytes) -> void* {
    void* p = wsp + off; off += (bytes + 255) & ~(size_t)255; return p;
  };
  float* z            = (float*)take((size_t)nMax * KN * 4);
  float* acc          = (float*)take((size_t)nMax * KN * 4);
  unsigned short* hbf = (unsigned short*)take((size_t)nMax * KN * 2);
  unsigned short* wbf = (unsigned short*)take((size_t)KN * KN * 2);
  float* Bbuf         = (float*)take((size_t)EXER * KN * 4);
  float* Cbuf         = (float*)take((size_t)EXER * KN * 4);
  float* s1           = (float*)take((size_t)nMax * 4);
  float* s2           = (float*)take((size_t)nMax * 4);
  unsigned* m_ord     = (unsigned*)take((size_t)nMax * 4);
  float* denom        = (float*)take((size_t)nMax * 4);

  float* out_kn   = (float*)d_out;
  float* out_exer = out_kn + (size_t)KN * KN;
  float* out_stu  = out_exer + (size_t)EXER * KN;

  const int TPB = 256;
  auto cdiv = [](long long a, long long b) { return (int)((a + b - 1) / b); };

  auto run_layer = [&](const float* W, const float* a, const int* src,
                       const int* dst, int nE, int nNodes) {
    cvt_concat_bf16<<<cdiv((long long)KN * KN, TPB), TPB, 0, stream>>>(
        wbf, W, KN, (const float*)nullptr, (size_t)KN * KN);
    gemm_bf16_wmma<<<dim3(cdiv(nNodes / 16, 8), KN / NHALF), TPB, 0, stream>>>(
        z, hbf, wbf, nNodes);
    row_dots<<<cdiv(nNodes, 8), TPB, 0, stream>>>(s1, s2, z, a, nNodes);
    init_layer<<<cdiv((long long)nNodes * KN, TPB), TPB, 0, stream>>>(
        m_ord, denom, acc, nNodes);
    edge_max<<<cdiv(nE, TPB), TPB, 0, stream>>>(src, dst, s1, s2, m_ord, nE);
    edge_sum<<<cdiv(nE, TPB), TPB, 0, stream>>>(src, dst, s1, s2, m_ord, denom, nE);
    edge_scatter<<<cdiv(nE, 8), TPB, 0, stream>>>(src, dst, s1, s2, m_ord,
                                                  denom, z, acc, nE);
  };

  // ===== exer<->knowledge graph: h = concat(exer_emb, kn_emb) =====
  cvt_concat_bf16<<<cdiv((long long)nEK * KN, TPB), TPB, 0, stream>>>(
      hbf, exer_emb, EXER, kn_emb, (size_t)nEK * KN);

  // layer ke: kn_out = kn_emb + D (score3 == 1 exactly)
  run_layer(W_ke, a_ke, src_ke, dst_ke, E_KE, nEK);
  add_rows<<<cdiv((long long)KN * KN, TPB), TPB, 0, stream>>>(
      out_kn, kn_emb, acc + (size_t)EXER * KN, (size_t)KN * KN);

  // layer ek: B = e_from_k[:EXER]
  run_layer(W_ek, a_ek, src_ek, dst_ek, E_KE, nEK);
  hipMemcpyAsync(Bbuf, acc, (size_t)EXER * KN * 4, hipMemcpyDeviceToDevice, stream);

  // ===== exer<->student graph: h = concat(exer_emb, stu_emb) =====
  cvt_concat_bf16<<<cdiv((long long)nES * KN, TPB), TPB, 0, stream>>>(
      hbf, exer_emb, EXER, stu_emb, (size_t)nES * KN);

  // layer se: stu_out = stu_emb + u_from_e[EXER:]
  run_layer(W_se, a_se, src_se, dst_se, E_SE, nES);
  add_rows<<<cdiv((long long)STU * KN, TPB), TPB, 0, stream>>>(
      out_stu, stu_emb, acc + (size_t)EXER * KN, (size_t)STU * KN);

  // layer es: C = e_from_u[:EXER]
  run_layer(W_es, a_es, src_es, dst_es, E_SE, nES);
  hipMemcpyAsync(Cbuf, acc, (size_t)EXER * KN * 4, hipMemcpyDeviceToDevice, stream);

  // ===== exer_out = emb + softmax(s1,s2) · (B,C) =====
  exer_final<<<cdiv(EXER, 8), TPB, 0, stream>>>(out_exer, exer_emb, Bbuf, Cbuf,
                                                e1_w, e1_b, e2_w, e2_b, EXER);
}
